// GCN_ensemble_89472758710374
// MI455X (gfx1250) — matrix operand order repaired and checked by reading, not compiled, not run on hardware
//
#include <hip/hip_runtime.h>
#include <hip/hip_bf16.h>

typedef __attribute__((ext_vector_type(16))) _Float16 v16h;
typedef __attribute__((ext_vector_type(8)))  float    v8f;

// ---------------------------------------------------------------- utilities

__global__ void zero_f32_kernel(float* __restrict__ p, size_t n) {
  size_t i = (size_t)blockIdx.x * blockDim.x + threadIdx.x;
  if (i < n) p[i] = 0.0f;
}

__global__ void f32_to_f16_kernel(const float* __restrict__ in,
                                  _Float16* __restrict__ out, size_t n) {
  size_t i = (size_t)blockIdx.x * blockDim.x + threadIdx.x;
  if (i < n) out[i] = (_Float16)in[i];
}

// deg[dst] += 1 for every edge (indices are int32 per harness ABI)
__global__ void deg_count_kernel(const int* __restrict__ ei, int E, float* __restrict__ deg) {
  int e = blockIdx.x * blockDim.x + threadIdx.x;
  if (e < E) atomicAdd(&deg[ei[E + e]], 1.0f);
}

// d = rsqrt(deg + 1)  (self loop included)
__global__ void rsqrt_kernel(float* __restrict__ d, int n) {
  int i = blockIdx.x * blockDim.x + threadIdx.x;
  if (i < n) d[i] = rsqrtf(d[i] + 1.0f);
}

// ------------------------------------------------------- WMMA GEMM + scale
// H[row,:] = ACC[row,:] = (A[row,:K] @ W[:K,:NC]) * dinv[row]
// One wave -> 16 rows x NC cols. Block: 256 thr = 8 waves = 128 rows.
template <int K, int NC>
__global__ __launch_bounds__(256)
void gemm_scale_kernel(const _Float16* __restrict__ A,
                       const _Float16* __restrict__ W,
                       const float* __restrict__ dinv,
                       float* __restrict__ H, float* __restrict__ ACC, int M) {
  __shared__ _Float16 Bl[K * NC];   // W transposed: Bl[n*K + k], 64 KB

  const int tid = threadIdx.x;
  for (int i = tid; i < K * NC; i += 256) {
    int k = i / NC, n = i % NC;
    Bl[n * K + k] = W[i];
  }
  __syncthreads();

  const int lane = tid & 31;
  const int wid  = tid >> 5;
  const int rm   = blockIdx.x * 128 + wid * 16;
  if (rm >= M) return;

  const int m  = lane & 15;   // row within tile (A), col within tile (B/D)
  const int hi = lane >> 4;   // lane-group select

  float dv[8];
#pragma unroll
  for (int r = 0; r < 8; ++r) {
    int row = rm + hi * 8 + r;
    dv[r] = dinv[row < M ? row : (M - 1)];
  }

  v8f acc[NC / 16] = {};

  int arow = rm + m; if (arow >= M) arow = M - 1;
  const _Float16* __restrict__ ap = A + (size_t)arow * K + hi * 16;

  for (int kb = 0; kb < K; kb += 32) {
    // A fragment, K contiguous per half (matches B):
    // lanes 0-15: K = kb..kb+15 ; lanes 16-31: K = kb+16..kb+31
    v16h a;
#pragma unroll
    for (int v = 0; v < 8; ++v) {
      int koff = v << 1;
      a[2 * v]     = ap[kb + koff];
      a[2 * v + 1] = ap[kb + koff + 1];
    }
#pragma unroll
    for (int t = 0; t < NC / 16; ++t) {
      // B fragment: lane holds column n = t*16+m, halves are K contiguous
      // (lanes 0-15 -> K=kb..kb+15, lanes 16-31 -> K=kb+16..kb+31)
      const _Float16* bp = Bl + (size_t)(t * 16 + m) * K + kb + hi * 16;
      v16h b = *(const v16h*)bp;
      acc[t] = __builtin_amdgcn_wmma_f32_16x16x32_f16(
          false, a, false, b, (short)0, acc[t], false, false);
    }
  }

  // D layout: lane L, vgpr r -> row rm + (L>=16?8:0) + r, col t*16 + (L&15)
#pragma unroll
  for (int t = 0; t < NC / 16; ++t) {
#pragma unroll
    for (int r = 0; r < 8; ++r) {
      int row = rm + hi * 8 + r;
      if (row < M) {
        float val = acc[t][r] * dv[r];
        size_t off = (size_t)row * NC + t * 16 + m;
        H[off]   = val;   // gather source (h' = h * dinv)
        ACC[off] = val;   // accumulator seeded with self-loop term
      }
    }
  }
}

// ------------------------------------------------------------ edge scatter
// ACC[dst,:] += H[src,:]   (C/4 threads per edge, float4 + 4 f32 atomics)
template <int C>
__global__ void scatter_add_kernel(const int* __restrict__ ei, int E,
                                   const float* __restrict__ H,
                                   float* __restrict__ ACC) {
  constexpr int TPE = C / 4;
  size_t gid = (size_t)blockIdx.x * blockDim.x + threadIdx.x;
  size_t e   = gid / TPE;
  int   c4   = (int)(gid % TPE) * 4;
  if (e >= (size_t)E) return;
  int src = ei[e];
  int dst = ei[(size_t)E + e];
  const float4 v = *(const float4*)(H + (size_t)src * C + c4);
  float* p = ACC + (size_t)dst * C + c4;
  atomicAdd(p + 0, v.x);
  atomicAdd(p + 1, v.y);
  atomicAdd(p + 2, v.z);
  atomicAdd(p + 3, v.w);
}

// --------------------------------------------------------------- epilogues
// hidden layer: T = f16(relu(ACC * dinv[row] + bias[col]))
__global__ void finish_relu_f16_kernel(const float* __restrict__ ACC,
                                       const float* __restrict__ dinv,
                                       const float* __restrict__ bias,
                                       _Float16* __restrict__ T, int M, int C) {
  size_t i = (size_t)blockIdx.x * blockDim.x + threadIdx.x;
  if (i >= (size_t)M * C) return;
  int row = (int)(i / C), c = (int)(i % C);
  float v = ACC[i] * dinv[row] + bias[c];
  T[i] = (_Float16)(v > 0.0f ? v : 0.0f);
}

// output layer: out (=|+=) ACC * dinv[row] + bias[col]
__global__ void finish_out_kernel(const float* __restrict__ ACC,
                                  const float* __restrict__ dinv,
                                  const float* __restrict__ bias,
                                  float* __restrict__ out, int M, int C,
                                  int accumulate) {
  size_t i = (size_t)blockIdx.x * blockDim.x + threadIdx.x;
  if (i >= (size_t)M * C) return;
  int row = (int)(i / C), c = (int)(i % C);
  float v = ACC[i] * dinv[row] + bias[c];
  out[i] = accumulate ? (out[i] + v) : v;
}

// ------------------------------------------------------------------ driver

static inline unsigned nblk(size_t n, int t) { return (unsigned)((n + t - 1) / t); }

extern "C" void kernel_launch(void* const* d_in, const int* in_sizes, int n_in,
                              void* d_out, int out_size, void* d_ws, size_t ws_size,
                              hipStream_t stream) {
  (void)n_in; (void)out_size; (void)ws_size;

  const float* X   = (const float*)d_in[0];
  const int*   eiA = (const int*)d_in[1];
  const int*   eiB = (const int*)d_in[2];
  const float* W0  = (const float*)d_in[3];
  const float* b0  = (const float*)d_in[4];
  const float* W1  = (const float*)d_in[5];
  const float* b1  = (const float*)d_in[6];
  const float* W2  = (const float*)d_in[7];
  const float* b2  = (const float*)d_in[8];
  const float* W3  = (const float*)d_in[9];
  const float* b3  = (const float*)d_in[10];
  float* out = (float*)d_out;

  const int IN = 128, HID = 256, OUT = 128;
  const int N = in_sizes[0] / IN;
  const int E = in_sizes[1] / 2;

  // workspace carve
  char* ws = (char*)d_ws;
  size_t off = 0;
  auto carve = [&](size_t bytes) -> char* {
    off = (off + 255) & ~(size_t)255;
    char* p = ws + off;
    off += bytes;
    return p;
  };
  _Float16* Xh   = (_Float16*)carve((size_t)N * IN * 2);
  _Float16* W0h  = (_Float16*)carve((size_t)IN * HID * 2);
  _Float16* W1h  = (_Float16*)carve((size_t)IN * HID * 2);
  _Float16* W2h  = (_Float16*)carve((size_t)HID * OUT * 2);
  _Float16* W3h  = (_Float16*)carve((size_t)HID * OUT * 2);
  float*    dinvA = (float*)carve((size_t)N * 4);
  float*    dinvB = (float*)carve((size_t)N * 4);
  float*    Hbuf  = (float*)carve((size_t)N * HID * 4);  // reused as N x 128 in layer 2
  float*    Abuf  = (float*)carve((size_t)N * HID * 4);  // reused as N x 128 in layer 2
  _Float16* Th    = (_Float16*)carve((size_t)N * HID * 2);

  // f16 conversions
  f32_to_f16_kernel<<<nblk((size_t)N * IN, 256), 256, 0, stream>>>(X, Xh, (size_t)N * IN);
  f32_to_f16_kernel<<<nblk((size_t)IN * HID, 256), 256, 0, stream>>>(W0, W0h, (size_t)IN * HID);
  f32_to_f16_kernel<<<nblk((size_t)IN * HID, 256), 256, 0, stream>>>(W1, W1h, (size_t)IN * HID);
  f32_to_f16_kernel<<<nblk((size_t)HID * OUT, 256), 256, 0, stream>>>(W2, W2h, (size_t)HID * OUT);
  f32_to_f16_kernel<<<nblk((size_t)HID * OUT, 256), 256, 0, stream>>>(W3, W3h, (size_t)HID * OUT);

  // degree -> dinv for both graphs
  zero_f32_kernel<<<nblk(N, 256), 256, 0, stream>>>(dinvA, (size_t)N);
  zero_f32_kernel<<<nblk(N, 256), 256, 0, stream>>>(dinvB, (size_t)N);
  deg_count_kernel<<<nblk(E, 256), 256, 0, stream>>>(eiA, E, dinvA);
  deg_count_kernel<<<nblk(E, 256), 256, 0, stream>>>(eiB, E, dinvB);
  rsqrt_kernel<<<nblk(N, 256), 256, 0, stream>>>(dinvA, N);
  rsqrt_kernel<<<nblk(N, 256), 256, 0, stream>>>(dinvB, N);

  const unsigned gblk = nblk((size_t)N, 128);

  // ---------------- branch 1: graph A ----------------
  gemm_scale_kernel<128, 256><<<gblk, 256, 0, stream>>>(Xh, W0h, dinvA, Hbuf, Abuf, N);
  scatter_add_kernel<256><<<nblk((size_t)E * 64, 256), 256, 0, stream>>>(eiA, E, Hbuf, Abuf);
  finish_relu_f16_kernel<<<nblk((size_t)N * 256, 256), 256, 0, stream>>>(Abuf, dinvA, b0, Th, N, 256);

  gemm_scale_kernel<256, 128><<<gblk, 256, 0, stream>>>(Th, W2h, dinvA, Hbuf, Abuf, N);
  scatter_add_kernel<128><<<nblk((size_t)E * 32, 256), 256, 0, stream>>>(eiA, E, Hbuf, Abuf);
  finish_out_kernel<<<nblk((size_t)N * 128, 256), 256, 0, stream>>>(Abuf, dinvA, b2, out, N, 128, 0);

  // ---------------- branch 2: graph B ----------------
  gemm_scale_kernel<128, 256><<<gblk, 256, 0, stream>>>(Xh, W1h, dinvB, Hbuf, Abuf, N);
  scatter_add_kernel<256><<<nblk((size_t)E * 64, 256), 256, 0, stream>>>(eiB, E, Hbuf, Abuf);
  finish_relu_f16_kernel<<<nblk((size_t)N * 256, 256), 256, 0, stream>>>(Abuf, dinvB, b1, Th, N, 256);

  gemm_scale_kernel<256, 128><<<gblk, 256, 0, stream>>>(Th, W3h, dinvB, Hbuf, Abuf, N);
  scatter_add_kernel<128><<<nblk((size_t)E * 32, 256), 256, 0, stream>>>(eiB, E, Hbuf, Abuf);
  finish_out_kernel<<<nblk((size_t)N * 128, 256), 256, 0, stream>>>(Abuf, dinvB, b3, out, N, 128, 1);
}
